// GPT_51084341018987
// MI455X (gfx1250) — compile-verified, matrix operand order checked by reading, not run on hardware
//
#include <hip/hip_runtime.h>
#include <math.h>

// ---------------------------------------------------------------------------
// Types for WMMA fragments (gfx1250, wave32)
// ---------------------------------------------------------------------------
typedef __bf16 v16bf __attribute__((ext_vector_type(16)));
typedef float  v8f   __attribute__((ext_vector_type(8)));
typedef unsigned int u32x4 __attribute__((ext_vector_type(4)));
typedef float  f32x4 __attribute__((ext_vector_type(4)));
typedef int    i32x4 __attribute__((ext_vector_type(4)));

union Frag {
  v16bf v;
  u32x4 q[2];
  unsigned short s[16];
};

__device__ __forceinline__ unsigned short f2bf(float f) {
  unsigned int u = __float_as_uint(f);
  unsigned int r = 0x7FFFu + ((u >> 16) & 1u);   // round-to-nearest-even
  return (unsigned short)((u + r) >> 16);
}

// ---------------------------------------------------------------------------
// Async global->LDS copy path (CDNA5 ASYNCcnt), guarded so compile never breaks
// Builtin signature (from hipcc diagnostic): arg0 = int4 __device__* (AS1),
// arg1 = int4 __shared__* (AS3), then two int immediates (offset, cpol).
// ---------------------------------------------------------------------------
#if defined(__has_builtin)
# if __has_builtin(__builtin_amdgcn_global_load_async_to_lds_b128) && \
     __has_builtin(__builtin_amdgcn_s_wait_asynccnt)
#  define USE_ASYNC_LDS 1
# endif
#endif
#ifndef USE_ASYNC_LDS
# define USE_ASYNC_LDS 0
#endif

__device__ __forceinline__ void copy16_g2l(const unsigned short* gsrc,
                                           unsigned short* ldst) {
#if USE_ASYNC_LDS
  __builtin_amdgcn_global_load_async_to_lds_b128(
      (__attribute__((address_space(1))) i32x4*)(void*)const_cast<unsigned short*>(gsrc),
      (__attribute__((address_space(3))) i32x4*)ldst,
      0, 0);
#else
  *(u32x4*)ldst = *(const u32x4*)gsrc;
#endif
}

__device__ __forceinline__ void async_wait_all() {
#if USE_ASYNC_LDS
  __builtin_amdgcn_s_wait_asynccnt(0);
#endif
}

#define OP_NONE 0
#define OP_RES  1
#define OP_GELU 2

// ---------------------------------------------------------------------------
// GEMM: C[M,N] = act( A[M,K](bf16) @ W[N,K]^T + bias ) (+ resid)
// Block tile 128x128, 8 waves (2 along M x 4 along N), wave tile 64x32
// = 4x2 WMMA 16x16 fragments = 8 v_wmma per 32-wide K-step.
// WBF16:  W already bf16 (pure async copy into LDS)
// OUTBF16: write bf16 output (feeds next WMMA consumer)
// ---------------------------------------------------------------------------
template<bool WBF16, bool OUTBF16>
__global__ __launch_bounds__(256) void gemm_kernel(
    const unsigned short* __restrict__ A,   // bf16 [M,K]
    const void* __restrict__ Wv,            // bf16 or f32 [N,K]
    const float* __restrict__ bias,
    const float* resid, void* Cv,
    int M, int N, int K, int op)
{
  __shared__ __align__(16) unsigned short As[128][40];  // +8 pad, rows 80B (16B-aligned)
  __shared__ __align__(16) unsigned short Bs[128][40];

  const int tid   = threadIdx.x;
  const int lane  = tid & 31, wid = tid >> 5;
  const int lhalf = lane >> 4, l16 = lane & 15;
  const int mBase = blockIdx.y * 128;
  const int nBase = blockIdx.x * 128;
  const int m0 = (wid & 1) * 64;     // 2 waves along M
  const int n0 = (wid >> 1) * 32;    // 4 waves along N

  v8f acc[4][2] = {};

  for (int k0 = 0; k0 < K; k0 += 32) {
    // A tile: 128x32 bf16 = 512 16B chunks, 2 per thread
#pragma unroll
    for (int it = 0; it < 2; ++it) {
      int chunk = it * 256 + tid;
      int r = chunk >> 2;
      int c = (chunk & 3) * 8;
      const unsigned short* g = A + (size_t)(mBase + r) * K + k0 + c;
      copy16_g2l(g, &As[r][c]);
      if (k0 + 32 < K) __builtin_prefetch(g + 32, 0, 0);
    }
    if (WBF16) {
      const unsigned short* Wb = (const unsigned short*)Wv;
#pragma unroll
      for (int it = 0; it < 2; ++it) {
        int chunk = it * 256 + tid;
        int r = chunk >> 2;
        int c = (chunk & 3) * 8;
        const unsigned short* g = Wb + (size_t)(nBase + r) * K + k0 + c;
        copy16_g2l(g, &Bs[r][c]);
        if (k0 + 32 < K) __builtin_prefetch(g + 32, 0, 0);
      }
    } else {
      const float* Wf = (const float*)Wv;
#pragma unroll
      for (int it = 0; it < 4; ++it) {              // 128x32 f32 -> bf16
        int chunk = it * 256 + tid;                 // 1024 float4 chunks
        int r = chunk >> 3;
        int c = (chunk & 7) * 4;
        const float* g = Wf + (size_t)(nBase + r) * K + k0 + c;
        f32x4 w4 = *(const f32x4*)g;
        Bs[r][c+0] = f2bf(w4.x); Bs[r][c+1] = f2bf(w4.y);
        Bs[r][c+2] = f2bf(w4.z); Bs[r][c+3] = f2bf(w4.w);
        if (k0 + 32 < K) __builtin_prefetch(g + 32, 0, 0);
      }
    }
    async_wait_all();
    __syncthreads();

    Frag af[4], bfr[2];
#pragma unroll
    for (int f = 0; f < 4; ++f) {
      // A 16x32 bf16 layout: lane<16 row=l16 K=0..7/16..23; lane>=16 K=8..15/24..31
      const unsigned short* pa = &As[m0 + f*16 + l16][lhalf * 8];
      af[f].q[0] = *(const u32x4*)pa;
      af[f].q[1] = *(const u32x4*)(pa + 16);
    }
#pragma unroll
    for (int f = 0; f < 2; ++f) {
      const unsigned short* pb = &Bs[n0 + f*16 + l16][lhalf * 8];
      bfr[f].q[0] = *(const u32x4*)pb;
      bfr[f].q[1] = *(const u32x4*)(pb + 16);
    }
#pragma unroll
    for (int fm = 0; fm < 4; ++fm)
#pragma unroll
      for (int fn = 0; fn < 2; ++fn)
        acc[fm][fn] = __builtin_amdgcn_wmma_f32_16x16x32_bf16(
            false, af[fm].v, false, bfr[fn].v, (short)0, acc[fm][fn], false, false);
    __syncthreads();
  }

  // Epilogue: C layout elem i -> m = i + 8*lhalf, n = l16
#pragma unroll
  for (int fm = 0; fm < 4; ++fm)
#pragma unroll
    for (int fn = 0; fn < 2; ++fn)
#pragma unroll
      for (int i = 0; i < 8; ++i) {
        int gm = mBase + m0 + fm * 16 + i + lhalf * 8;
        int gn = nBase + n0 + fn * 16 + l16;
        float v = acc[fm][fn][i];
        if (bias) v += bias[gn];
        if (op == OP_GELU) {
          float t = 0.7978845608028654f * (v + 0.044715f * v * v * v);
          v = 0.5f * v * (1.0f + tanhf(t));
        }
        if (op == OP_RES) v += resid[(size_t)gm * N + gn];
        if (OUTBF16) ((unsigned short*)Cv)[(size_t)gm * N + gn] = f2bf(v);
        else         ((float*)Cv)[(size_t)gm * N + gn] = v;
      }
}

// ---------------------------------------------------------------------------
// Fused causal flash-attention on bf16 qkv. Grid (T/64, B*H), 128 thr = 4 waves.
// Wave = 16 query rows; both QK^T and PV on v_wmma_f32_16x16x32_bf16.
// ---------------------------------------------------------------------------
__global__ __launch_bounds__(128) void attn_kernel(
    const unsigned short* __restrict__ qkv,   // bf16 [4096, 2304]
    unsigned short* __restrict__ y)           // bf16 [4096, 768]
{
  const int qblk = blockIdx.x;      // 16 q-blocks of 64 rows
  const int bh   = blockIdx.y;      // B*H = 48
  const int b = bh / 12, h = bh % 12;
  const int tid   = threadIdx.x;
  const int lane  = tid & 31, wid = tid >> 5;
  const int lhalf = lane >> 4, l16 = lane & 15;
  const int qwave = qblk * 64 + wid * 16;

  __shared__ __align__(16) unsigned short Ks [32][72];   // [key][dim]
  __shared__ __align__(16) unsigned short VsT[64][40];   // [dim][key] (transposed)
  __shared__ __align__(16) unsigned short Ps [4][16][40]; // per-wave P re-layout

  // Q fragments: 16x64 bf16 = 2 A-frags, direct b128 loads from global
  Frag qf[2];
  {
    const unsigned short* qp =
        qkv + (size_t)(b * 1024 + qwave + l16) * 2304 + h * 64;
#pragma unroll
    for (int ch = 0; ch < 2; ++ch) {
      qf[ch].q[0] = *(const u32x4*)(qp + ch * 32 + lhalf * 8);
      qf[ch].q[1] = *(const u32x4*)(qp + ch * 32 + 16 + lhalf * 8);
    }
  }

  float mrow[8], lrow[8];
  v8f oacc[4] = {};
#pragma unroll
  for (int i = 0; i < 8; ++i) { mrow[i] = -INFINITY; lrow[i] = 0.f; }

  const int nkt = qblk * 2 + 2;                 // uniform across block (causal)
  for (int kt = 0; kt < nkt; ++kt) {
    const int kbase = kt * 32;
    // K tile: 32x64 bf16 = 256 16B chunks, 2 per thread (pure copy -> async)
#pragma unroll
    for (int it = 0; it < 2; ++it) {
      int chunk = it * 128 + tid;
      int r = chunk >> 3;
      int c = (chunk & 7) * 8;
      copy16_g2l(qkv + (size_t)(b * 1024 + kbase + r) * 2304 + 768 + h * 64 + c,
                 &Ks[r][c]);
    }
    // V tile: load 8 halves, scatter transposed
#pragma unroll
    for (int it = 0; it < 2; ++it) {
      int chunk = it * 128 + tid;
      int r = chunk >> 3;
      int c = (chunk & 7) * 8;
      union { u32x4 q; unsigned short s[8]; } vv;
      vv.q = *(const u32x4*)(qkv + (size_t)(b * 1024 + kbase + r) * 2304 + 1536 +
                             h * 64 + c);
#pragma unroll
      for (int j = 0; j < 8; ++j) VsT[c + j][r] = vv.s[j];
    }
    async_wait_all();
    __syncthreads();

    // S = Q K^T  (2 n-frags of 16 keys, K-dim 64 = 2 wmma each)
    v8f sf[2];
#pragma unroll
    for (int nf = 0; nf < 2; ++nf) {
      Frag kb0, kb1;
      const unsigned short* pk = &Ks[nf * 16 + l16][lhalf * 8];
      kb0.q[0] = *(const u32x4*)pk;
      kb0.q[1] = *(const u32x4*)(pk + 16);
      kb1.q[0] = *(const u32x4*)(pk + 32);
      kb1.q[1] = *(const u32x4*)(pk + 48);
      v8f z = {};
      z = __builtin_amdgcn_wmma_f32_16x16x32_bf16(false, qf[0].v, false, kb0.v, (short)0, z, false, false);
      z = __builtin_amdgcn_wmma_f32_16x16x32_bf16(false, qf[1].v, false, kb1.v, (short)0, z, false, false);
      sf[nf] = z;
    }

    // causal mask + scale (1/sqrt(64))
#pragma unroll
    for (int nf = 0; nf < 2; ++nf)
#pragma unroll
      for (int i = 0; i < 8; ++i) {
        int qg = qwave + i + lhalf * 8;
        int kg = kbase + nf * 16 + l16;
        float s = sf[nf][i];
        sf[nf][i] = (kg <= qg) ? s * 0.125f : -INFINITY;
      }

    // online softmax; row = 16 lanes of one half-wave -> xor-shuffle reduce
#pragma unroll
    for (int i = 0; i < 8; ++i) {
      float cm = fmaxf(sf[0][i], sf[1][i]);
#pragma unroll
      for (int sh = 8; sh >= 1; sh >>= 1)
        cm = fmaxf(cm, __shfl_xor(cm, sh, 32));
      float mn = fmaxf(mrow[i], cm);
      float alpha = (mn == -INFINITY) ? 1.0f : __expf(mrow[i] - mn);
      float p0 = (sf[0][i] == -INFINITY) ? 0.f : __expf(sf[0][i] - mn);
      float p1 = (sf[1][i] == -INFINITY) ? 0.f : __expf(sf[1][i] - mn);
      sf[0][i] = p0; sf[1][i] = p1;
      float ps = p0 + p1;
#pragma unroll
      for (int sh = 8; sh >= 1; sh >>= 1)
        ps += __shfl_xor(ps, sh, 32);
      lrow[i] = lrow[i] * alpha + ps;
      mrow[i] = mn;
#pragma unroll
      for (int df = 0; df < 4; ++df)
        oacc[df][i] *= alpha;
    }

    // C-frag P -> LDS -> A-frag P (per-wave buffer; wave-local LDS is in-order)
#pragma unroll
    for (int nf = 0; nf < 2; ++nf)
#pragma unroll
      for (int i = 0; i < 8; ++i)
        Ps[wid][i + lhalf * 8][nf * 16 + l16] = f2bf(sf[nf][i]);

    Frag pf;
    const unsigned short* pp = &Ps[wid][l16][lhalf * 8];
    pf.q[0] = *(const u32x4*)pp;
    pf.q[1] = *(const u32x4*)(pp + 16);

    // O += P[16x32] @ V[32x64]  (4 dim-frags)
#pragma unroll
    for (int df = 0; df < 4; ++df) {
      Frag vf;
      const unsigned short* pv = &VsT[df * 16 + l16][lhalf * 8];
      vf.q[0] = *(const u32x4*)pv;
      vf.q[1] = *(const u32x4*)(pv + 16);
      oacc[df] = __builtin_amdgcn_wmma_f32_16x16x32_bf16(
          false, pf.v, false, vf.v, (short)0, oacc[df], false, false);
    }
    __syncthreads();
  }

  // normalize and scatter back to [B,T,E] in bf16
#pragma unroll
  for (int i = 0; i < 8; ++i) {
    float inv = 1.0f / lrow[i];
    int qg = qwave + i + lhalf * 8;
#pragma unroll
    for (int df = 0; df < 4; ++df) {
      int dg = h * 64 + df * 16 + l16;
      y[(size_t)(b * 1024 + qg) * 768 + dg] = f2bf(oacc[df][i] * inv);
    }
  }
}

// ---------------------------------------------------------------------------
// LayerNorm: one row (768) per block; fp32 in, bf16 out (feeds WMMA A)
// ---------------------------------------------------------------------------
__global__ __launch_bounds__(256) void ln_kernel(const float* __restrict__ x,
    const float* __restrict__ g, const float* __restrict__ bta,
    unsigned short* __restrict__ out)
{
  __shared__ float red[256];
  const int row = blockIdx.x;
  const int tid = threadIdx.x;
  const float* xr = x + (size_t)row * 768;
  float s = 0.f;
  for (int e = tid; e < 768; e += 256) s += xr[e];
  red[tid] = s; __syncthreads();
  for (int off = 128; off > 0; off >>= 1) {
    if (tid < off) red[tid] += red[tid + off];
    __syncthreads();
  }
  float mean = red[0] * (1.0f / 768.0f);
  __syncthreads();
  float v = 0.f;
  for (int e = tid; e < 768; e += 256) { float d = xr[e] - mean; v += d * d; }
  red[tid] = v; __syncthreads();
  for (int off = 128; off > 0; off >>= 1) {
    if (tid < off) red[tid] += red[tid + off];
    __syncthreads();
  }
  float rstd = rsqrtf(red[0] * (1.0f / 768.0f) + 1e-5f);
  for (int e = tid; e < 768; e += 256)
    out[(size_t)row * 768 + e] = f2bf((xr[e] - mean) * rstd * g[e] + bta[e]);
}

// ---------------------------------------------------------------------------
// Weight fp32 -> bf16 staging (each weight element converted exactly once)
// ---------------------------------------------------------------------------
__global__ __launch_bounds__(256) void wcvt_kernel(const float* __restrict__ src,
    unsigned short* __restrict__ dst, int n)
{
  int i = (blockIdx.x * 256 + threadIdx.x) * 4;
  if (i < n) {
    f32x4 v = *(const f32x4*)(src + i);
    dst[i+0] = f2bf(v.x); dst[i+1] = f2bf(v.y);
    dst[i+2] = f2bf(v.z); dst[i+3] = f2bf(v.w);
  }
}

// ---------------------------------------------------------------------------
// Token + position embedding (fp32 residual trunk)
// ---------------------------------------------------------------------------
__global__ __launch_bounds__(256) void embed_kernel(const int* __restrict__ idx,
    const float* __restrict__ wte, const float* __restrict__ wpe,
    float* __restrict__ x)
{
  const int bt = blockIdx.x;          // 0..4095
  const int t = bt & 1023;
  const int tok = idx[bt];
  for (int e = threadIdx.x; e < 768; e += 256)
    x[(size_t)bt * 768 + e] = wte[(size_t)tok * 768 + e] + wpe[(size_t)t * 768 + e];
}

// ---------------------------------------------------------------------------
// Cross-entropy: per-row log-softmax loss, then deterministic mean reduce
// ---------------------------------------------------------------------------
__global__ __launch_bounds__(256) void loss_row_kernel(const float* __restrict__ logits,
    const int* __restrict__ targets, float* __restrict__ rowloss)
{
  __shared__ float red[256];
  const int r = blockIdx.x;
  const int tid = threadIdx.x;
  const float* lr = logits + (size_t)r * 50304;
  float mx = -INFINITY;
  for (int i = tid; i < 50304; i += 256) mx = fmaxf(mx, lr[i]);
  red[tid] = mx; __syncthreads();
  for (int off = 128; off > 0; off >>= 1) {
    if (tid < off) red[tid] = fmaxf(red[tid], red[tid + off]);
    __syncthreads();
  }
  mx = red[0]; __syncthreads();
  float s = 0.f;
  for (int i = tid; i < 50304; i += 256) s += __expf(lr[i] - mx);
  red[tid] = s; __syncthreads();
  for (int off = 128; off > 0; off >>= 1) {
    if (tid < off) red[tid] += red[tid + off];
    __syncthreads();
  }
  if (tid == 0)
    rowloss[r] = -(lr[targets[r]] - mx - __logf(red[0]));
}

__global__ __launch_bounds__(256) void loss_reduce_kernel(
    const float* __restrict__ rowloss, float* __restrict__ out)
{
  __shared__ float red[256];
  const int tid = threadIdx.x;
  float s = 0.f;
  for (int i = tid; i < 4096; i += 256) s += rowloss[i];
  red[tid] = s; __syncthreads();
  for (int off = 128; off > 0; off >>= 1) {
    if (tid < off) red[tid] += red[tid + off];
    __syncthreads();
  }
  if (tid == 0) out[0] = red[0] * (1.0f / 4096.0f);
}

// ---------------------------------------------------------------------------
// Host orchestration
// ---------------------------------------------------------------------------
extern "C" void kernel_launch(void* const* d_in, const int* in_sizes, int n_in,
                              void* d_out, int out_size, void* d_ws, size_t ws_size,
                              hipStream_t stream)
{
  (void)in_sizes; (void)n_in; (void)ws_size;
  const int*   idx     = (const int*)d_in[0];
  const int*   targets = (const int*)d_in[1];
  const float* wte     = (const float*)d_in[2];
  const float* wpe     = (const float*)d_in[3];
  const float* ln1_g   = (const float*)d_in[4];
  const float* ln1_b   = (const float*)d_in[5];
  const float* attn_w  = (const float*)d_in[6];
  const float* attn_b  = (const float*)d_in[7];
  const float* proj_w  = (const float*)d_in[8];
  const float* proj_b  = (const float*)d_in[9];
  const float* ln2_g   = (const float*)d_in[10];
  const float* ln2_b   = (const float*)d_in[11];
  const float* fc_w    = (const float*)d_in[12];
  const float* fc_b    = (const float*)d_in[13];
  const float* fc2_w   = (const float*)d_in[14];
  const float* fc2_b   = (const float*)d_in[15];
  const float* lnf_g   = (const float*)d_in[16];
  const float* lnf_b   = (const float*)d_in[17];
  const float* lm_w    = (const float*)d_in[18];

  const int M = 4096;                                   // B*T
  const size_t ME = (size_t)M * 768;

  float* x = (float*)d_ws;                              // fp32 trunk [4096,768]
  unsigned short* hb   = (unsigned short*)(x + ME);     // bf16 ln output
  unsigned short* yb   = hb + ME;                       // bf16 attn output
  unsigned short* wbuf = yb + ME;                       // bf16 weight staging (<=2.36M elems)
  float* rowloss = (float*)(wbuf + (size_t)3072 * 768); // [4096]

  float* logits = (float*)d_out;
  // Transient bf16 buffers inside the (huge) logits region; both are dead
  // before the final LM-head GEMM overwrites all of d_out.
  unsigned short* qkvb = (unsigned short*)d_out;                          // [4096,2304]
  unsigned short* fcb  = (unsigned short*)d_out + (size_t)16 * 1024 * 1024; // [4096,3072]

  embed_kernel<<<M, 256, 0, stream>>>(idx, wte, wpe, x);

  for (int l = 0; l < 12; ++l) {
    ln_kernel<<<M, 256, 0, stream>>>(x, ln1_g + (size_t)l*768, ln1_b + (size_t)l*768, hb);
    wcvt_kernel<<<1728, 256, 0, stream>>>(attn_w + (size_t)l*2304*768, wbuf, 2304*768);
    gemm_kernel<true, true><<<dim3(18, 32), 256, 0, stream>>>(
        hb, wbuf, attn_b + (size_t)l*2304, nullptr, qkvb, M, 2304, 768, OP_NONE);
    attn_kernel<<<dim3(16, 48), 128, 0, stream>>>(qkvb, yb);
    wcvt_kernel<<<576, 256, 0, stream>>>(proj_w + (size_t)l*768*768, wbuf, 768*768);
    gemm_kernel<true, false><<<dim3(6, 32), 256, 0, stream>>>(
        yb, wbuf, proj_b + (size_t)l*768, x, x, M, 768, 768, OP_RES);
    ln_kernel<<<M, 256, 0, stream>>>(x, ln2_g + (size_t)l*768, ln2_b + (size_t)l*768, hb);
    wcvt_kernel<<<2304, 256, 0, stream>>>(fc_w + (size_t)l*3072*768, wbuf, 3072*768);
    gemm_kernel<true, true><<<dim3(24, 32), 256, 0, stream>>>(
        hb, wbuf, fc_b + (size_t)l*3072, nullptr, fcb, M, 3072, 768, OP_GELU);
    wcvt_kernel<<<2304, 256, 0, stream>>>(fc2_w + (size_t)l*768*3072, wbuf, 768*3072);
    gemm_kernel<true, false><<<dim3(6, 32), 256, 0, stream>>>(
        fcb, wbuf, fc2_b + (size_t)l*768, x, x, M, 768, 3072, OP_RES);
  }

  ln_kernel<<<M, 256, 0, stream>>>(x, lnf_g, lnf_b, hb);
  gemm_kernel<false, false><<<dim3(393, 32), 256, 0, stream>>>(
      hb, lm_w, nullptr, nullptr, logits, M, 50304, 768, OP_NONE);
  loss_row_kernel<<<M, 256, 0, stream>>>(logits, targets, rowloss);
  loss_reduce_kernel<<<1, 256, 0, stream>>>(rowloss, (float*)d_out + ((size_t)out_size - 1));
}
